// Encoder_25598005084923
// MI455X (gfx1250) — matrix operand order repaired and checked by reading, not compiled, not run on hardware
//
#include <hip/hip_runtime.h>
#include <hip/hip_bf16.h>
#include <math.h>

// Problem constants (match reference)
#define BATCH 64
#define TT    256
#define EMBD  256
#define HH    512
#define G4H   2048          // 4*H
#define BT    (BATCH*TT)    // 16384

typedef __bf16 bf16;
typedef __bf16 v16bf __attribute__((ext_vector_type(16)));
typedef float  v8f   __attribute__((ext_vector_type(8)));

union V16 { v16bf v; uint4 q[2]; bf16 e[16]; };

__device__ inline v8f zero8() {
  v8f z;
#pragma unroll
  for (int i = 0; i < 8; ++i) z[i] = 0.f;
  return z;
}

__device__ inline v8f wmma_bf16(v16bf a, v16bf b, v8f c) {
  // D = A(16x32 bf16) * B(32x16 bf16) + C(f32)  -> v_wmma_f32_16x16x32_bf16
  return __builtin_amdgcn_wmma_f32_16x16x32_bf16(
      /*neg_a=*/false, a, /*neg_b=*/false, b,
      /*c_mod=*/(short)0, c, /*reuse_a=*/false, /*reuse_b=*/false);
}

// ---------------- workspace layout (bytes) ----------------
// ~281 MB total.
#define WS_CNT   0u                                  // barrier counter (256 B)
#define WS_WHF   256ull                              // Wh_f  bf16  [2048*512]  2 MB
#define WS_WHB   (WS_WHF  + 2ull*1024*1024)
#define WS_WIF   (WS_WHB  + 2ull*1024*1024)
#define WS_WIB   (WS_WIF  + 2ull*1024*1024)
#define WS_WRES  (WS_WIB  + 2ull*1024*1024)          // W_res bf16 [256*2048] 1 MB
#define WS_HBF   (WS_WRES + 1ull*1024*1024)          // h ping-pong [2dir][2buf][64*512] bf16 256 KB
#define WS_EMBBF (WS_HBF  + 256ull*1024)             // embedded bf16 [16384*512] 16 MB
#define WS_XWF   (WS_EMBBF + 16ull*1024*1024)        // xw fwd f32 [16384*2048] 128 MB
#define WS_XWB   (WS_XWF  + 128ull*1024*1024)        // xw bwd f32 128 MB

// ---------------------------------------------------------------------------
// K0: zero outputs region, zero barrier + h ping-pong, convert weights to bf16
// ---------------------------------------------------------------------------
__global__ __launch_bounds__(256) void k_init(
    float* out_zero, const float* WhF, const float* WhB, const float* WiF,
    const float* WiB, const float* Wres, unsigned char* ws) {
  bf16* whf  = (bf16*)(ws + WS_WHF);
  bf16* whb  = (bf16*)(ws + WS_WHB);
  bf16* wif  = (bf16*)(ws + WS_WIF);
  bf16* wib  = (bf16*)(ws + WS_WIB);
  bf16* wres = (bf16*)(ws + WS_WRES);
  bf16* hbf  = (bf16*)(ws + WS_HBF);
  size_t i0 = (size_t)blockIdx.x * blockDim.x + threadIdx.x;
  size_t stride = (size_t)gridDim.x * blockDim.x;
  for (size_t i = i0; i < 16777216ull; i += stride) {  // outputs [B,T,2H]
    out_zero[i] = 0.f;
    if (i < 1048576ull) {                 // 2048*512 weight elements x4
      whf[i] = (bf16)WhF[i]; whb[i] = (bf16)WhB[i];
      wif[i] = (bf16)WiF[i]; wib[i] = (bf16)WiB[i];
    } else if (i < 1048576ull + 524288ull) {
      size_t j = i - 1048576ull; wres[j] = (bf16)Wres[j];
    } else if (i < 1048576ull + 524288ull + 131072ull) {
      hbf[i - 1572864ull] = (bf16)0.f;    // both ping-pong buffers, both dirs
    }
  }
  if (blockIdx.x == 0 && threadIdx.x < 64)
    ((unsigned*)(ws + WS_CNT))[threadIdx.x] = 0u;
}

// ---------------------------------------------------------------------------
// K1: embedding gather (cols 0..255 of embedded): act vs sum-of-obs
// ---------------------------------------------------------------------------
__global__ __launch_bounds__(256) void k_embed(
    const int* __restrict__ mask, const int* __restrict__ act_ids,
    const int* __restrict__ obs_ids, const float* __restrict__ act_emb,
    const float* __restrict__ obs_emb, float* __restrict__ out_emb,
    bf16* __restrict__ emb_bf) {
  size_t i0 = (size_t)blockIdx.x * blockDim.x + threadIdx.x;
  size_t stride = (size_t)gridDim.x * blockDim.x;
  for (size_t i = i0; i < (size_t)BT * EMBD; i += stride) {
    int bt = (int)(i >> 8), c = (int)(i & 255);
    float v;
    if (mask[bt]) {
      int id = act_ids[bt];
      v = id ? act_emb[id * EMBD + c] : 0.f;   // padding_idx 0
    } else {
      v = 0.f;
#pragma unroll
      for (int k = 0; k < 3; ++k) {
        int id = obs_ids[bt * 3 + k];
        if (id) v += obs_emb[id * EMBD + c];
      }
    }
    out_emb[(size_t)bt * 512 + c] = v;
    emb_bf[(size_t)bt * 512 + c] = (bf16)v;
  }
}

// ---------------------------------------------------------------------------
// K2: res projection GEMM.  rows=BT, N=256, K=4096 (sum over KR folded into K,
// B wraps mod 2048).  WG tile 64x128, wave owns one 16-col stripe, 4 m-tiles.
// ---------------------------------------------------------------------------
__global__ __launch_bounds__(256) void k_res(
    const float* __restrict__ res, const bf16* __restrict__ wres,
    const float* __restrict__ bres, const int* __restrict__ mask,
    float* __restrict__ out_emb, bf16* __restrict__ emb_bf) {
  int rowblk = blockIdx.x >> 1, colblk = blockIdx.x & 1;
  int wave = threadIdx.x >> 5, lane = threadIdx.x & 31;
  int lh = lane >> 4, l16 = lane & 15;
  int gp = colblk * 128 + wave * 16 + l16;                 // output col (PROJ)
  const bf16* bptr = wres + (size_t)gp * 2048 + lh * 16;
  const float* aptr[4];
#pragma unroll
  for (int mi = 0; mi < 4; ++mi) {
    int r = rowblk * 64 + mi * 16 + l16;
    aptr[mi] = res + (size_t)r * 4096 + lh * 16;
  }
  v8f acc[4];
#pragma unroll
  for (int mi = 0; mi < 4; ++mi) acc[mi] = zero8();

  for (int kc = 0; kc < 128; ++kc) {
    int kb = kc * 32;
    v16bf bf_ = *(const v16bf*)(bptr + (kb & 2047));       // wrap mod 2048
#pragma unroll
    for (int mi = 0; mi < 4; ++mi) {
      const float* p = aptr[mi] + kb;
      __builtin_prefetch(p + 256, 0, 0);                   // global_prefetch_b8
      float4 f0 = *(const float4*)p;
      float4 f1 = *(const float4*)(p + 4);
      float4 f2 = *(const float4*)(p + 8);
      float4 f3 = *(const float4*)(p + 12);
      V16 a;
      a.e[0]=(bf16)f0.x; a.e[1]=(bf16)f0.y; a.e[2]=(bf16)f0.z; a.e[3]=(bf16)f0.w;
      a.e[4]=(bf16)f1.x; a.e[5]=(bf16)f1.y; a.e[6]=(bf16)f1.z; a.e[7]=(bf16)f1.w;
      a.e[8]=(bf16)f2.x; a.e[9]=(bf16)f2.y; a.e[10]=(bf16)f2.z; a.e[11]=(bf16)f2.w;
      a.e[12]=(bf16)f3.x; a.e[13]=(bf16)f3.y; a.e[14]=(bf16)f3.z; a.e[15]=(bf16)f3.w;
      acc[mi] = wmma_bf16(a.v, bf_, acc[mi]);
    }
  }
  float bias = bres[gp];
#pragma unroll
  for (int mi = 0; mi < 4; ++mi)
#pragma unroll
    for (int j = 0; j < 8; ++j) {
      int bt = rowblk * 64 + mi * 16 + j + lh * 8;
      float v = mask[bt] ? 0.f : (acc[mi][j] + bias);
      out_emb[(size_t)bt * 512 + 256 + gp] = v;
      emb_bf[(size_t)bt * 512 + 256 + gp] = (bf16)v;
    }
}

// ---------------------------------------------------------------------------
// K3: input projection xw = x(_rev) @ Wi^T + (bi+bh).  M=16384, N=2048, K=512.
// backward gathers A rows via t = clip(len[b]-1-s, 0).
// ---------------------------------------------------------------------------
__global__ __launch_bounds__(256) void k_xw(
    const bf16* __restrict__ emb_bf, const bf16* __restrict__ wi,
    const float* __restrict__ bi, const float* __restrict__ bh,
    const int* __restrict__ lens, float* __restrict__ xw, int backward) {
  int rowblk = blockIdx.x >> 4;            // 0..255
  int colblk = blockIdx.x & 15;            // 0..15
  int wave = threadIdx.x >> 5, lane = threadIdx.x & 31;
  int lh = lane >> 4, l16 = lane & 15;
  int gcol = colblk * 128 + wave * 16 + l16;               // 0..2047
  const bf16* bptr = wi + (size_t)gcol * 512 + lh * 16;
  const bf16* aptr[4];
#pragma unroll
  for (int mi = 0; mi < 4; ++mi) {
    int r = rowblk * 64 + mi * 16 + l16;
    int src = r;
    if (backward) {
      int b = r >> 8, s = r & 255;
      int t = lens[b] - 1 - s; if (t < 0) t = 0;
      src = (b << 8) + t;
    }
    aptr[mi] = emb_bf + (size_t)src * 512 + lh * 16;
  }
  v8f acc[4];
#pragma unroll
  for (int mi = 0; mi < 4; ++mi) acc[mi] = zero8();
#pragma unroll
  for (int kc = 0; kc < 16; ++kc) {
    int kb = kc * 32;
    v16bf bf_ = *(const v16bf*)(bptr + kb);
#pragma unroll
    for (int mi = 0; mi < 4; ++mi) {
      V16 a;
      a.q[0] = *(const uint4*)(aptr[mi] + kb);
      a.q[1] = *(const uint4*)(aptr[mi] + kb + 8);
      acc[mi] = wmma_bf16(a.v, bf_, acc[mi]);
    }
  }
  float bias = bi[gcol] + bh[gcol];
#pragma unroll
  for (int mi = 0; mi < 4; ++mi)
#pragma unroll
    for (int j = 0; j < 8; ++j) {
      int r = rowblk * 64 + mi * 16 + j + lh * 8;
      xw[(size_t)r * 2048 + gcol] = acc[mi][j] + bias;
    }
}

// ---------------------------------------------------------------------------
// K4: persistent bidirectional LSTM scan.  grid = 64 WGs (2 dirs x 32 slices).
// Each WG owns 16 H-columns; each wave keeps its 512-deep Wh B-fragments in
// registers for all 256 steps; h ping-pongs in bf16 through global; gates are
// exchanged through LDS; one grid barrier per step.
// ---------------------------------------------------------------------------
__global__ __launch_bounds__(256) void k_lstm(
    const bf16* __restrict__ whf, const bf16* __restrict__ whb,
    const float* __restrict__ xwf, const float* __restrict__ xwb,
    const int* __restrict__ lens, bf16* __restrict__ hbf_base,
    float* __restrict__ out, float* __restrict__ out_h,
    float* __restrict__ out_c, unsigned* __restrict__ counter) {
  __shared__ float gate_s[4][64][16];   // 16 KB: i/f/g/o tiles for this slice

  int dir = blockIdx.x >> 5;
  int wg  = blockIdx.x & 31;
  int tid = threadIdx.x;
  int wave = tid >> 5, lane = tid & 31;
  int lh = lane >> 4, l16 = lane & 15;
  int q = wave >> 1, halfm = wave & 1;                     // gate, m-tile pair

  const bf16* wh = dir ? whb : whf;
  const float* xw = dir ? xwb : xwf;
  bf16* hbuf = hbf_base + (size_t)dir * 2 * (64 * 512);

  // Resident Wh B-fragments: N=16 cols of gate q, K=512 (16 chunks of 32).
  int gcol = q * 512 + wg * 16 + l16;
  v16bf Bf[16];
#pragma unroll
  for (int kc = 0; kc < 16; ++kc)
    Bf[kc] = *(const v16bf*)(wh + (size_t)gcol * 512 + kc * 32 + lh * 16);

  // Elementwise ownership: thread -> 4 (b, col) pairs, col fixed.
  int ecol = wg * 16 + (tid & 15);
  int eb[4], elen[4];
  float c4[4], hl[4];
#pragma unroll
  for (int p = 0; p < 4; ++p) {
    eb[p] = p * 16 + (tid >> 4);
    elen[p] = lens[eb[p]];
    c4[p] = 0.f; hl[p] = 0.f;
  }

  int r0 = halfm * 32 + l16;   // batch rows of this wave's two 16x16 m-tiles
  int r1 = r0 + 16;

  for (int s = 0; s < TT; ++s) {
    const bf16* hprev = hbuf + (s & 1) * (64 * 512);
    bf16* hnext = hbuf + ((s + 1) & 1) * (64 * 512);

    v8f acc0 = zero8(), acc1 = zero8();
    const bf16* ap0 = hprev + (size_t)r0 * 512 + lh * 16;
    const bf16* ap1 = hprev + (size_t)r1 * 512 + lh * 16;
#pragma unroll
    for (int kc = 0; kc < 16; ++kc) {
      int kb = kc * 32;
      V16 a0, a1;
      a0.q[0] = *(const uint4*)(ap0 + kb);
      a0.q[1] = *(const uint4*)(ap0 + kb + 8);
      a1.q[0] = *(const uint4*)(ap1 + kb);
      a1.q[1] = *(const uint4*)(ap1 + kb + 8);
      acc0 = wmma_bf16(a0.v, Bf[kc], acc0);
      acc1 = wmma_bf16(a1.v, Bf[kc], acc1);
    }
#pragma unroll
    for (int j = 0; j < 8; ++j) {
      gate_s[q][halfm * 32 + j + lh * 8][l16]      = acc0[j];
      gate_s[q][halfm * 32 + 16 + j + lh * 8][l16] = acc1[j];
    }
    __syncthreads();

#pragma unroll
    for (int p = 0; p < 4; ++p) {
      int b = eb[p];
      size_t xbase = ((size_t)(b * TT + s)) * 2048 + ecol;
      float gi = gate_s[0][b][tid & 15] + xw[xbase];
      float gf = gate_s[1][b][tid & 15] + xw[xbase + 512];
      float gg = gate_s[2][b][tid & 15] + xw[xbase + 1024];
      float go = gate_s[3][b][tid & 15] + xw[xbase + 1536];
      if (s < elen[p]) {
        float i_ = 1.f / (1.f + __expf(-gi));
        float f_ = 1.f / (1.f + __expf(-gf));
        float g_ = tanhf(gg);
        float o_ = 1.f / (1.f + __expf(-go));
        float cn = f_ * c4[p] + i_ * g_;
        float hn = o_ * tanhf(cn);
        c4[p] = cn; hl[p] = hn;
        hnext[b * 512 + ecol] = (bf16)hn;
        if (dir == 0) {
          out[((size_t)(b * TT + s)) * 1024 + ecol] = hn;
        } else {
          int t = elen[p] - 1 - s;
          out[((size_t)(b * TT + t)) * 1024 + 512 + ecol] = hn;
        }
      } else {
        hnext[b * 512 + ecol] = hprev[b * 512 + ecol];     // frozen state
      }
    }
    __syncthreads();
    __threadfence();                                       // release h_next
    if (tid == 0) {
      atomicAdd(counter, 1u);
      unsigned target = (unsigned)(s + 1) * 64u;
      while (__hip_atomic_load(counter, __ATOMIC_RELAXED,
                               __HIP_MEMORY_SCOPE_AGENT) < target)
        __builtin_amdgcn_s_sleep(1);
    }
    __syncthreads();
    __threadfence();                                       // acquire h_prev
  }

#pragma unroll
  for (int p = 0; p < 4; ++p) {
    out_h[(size_t)eb[p] * 1024 + dir * 512 + ecol] = hl[p];
    out_c[(size_t)eb[p] * 1024 + dir * 512 + ecol] = c4[p];
  }
}

// ---------------------------------------------------------------------------
extern "C" void kernel_launch(void* const* d_in, const int* in_sizes, int n_in,
                              void* d_out, int out_size, void* d_ws,
                              size_t ws_size, hipStream_t stream) {
  const int*   mask    = (const int*)d_in[0];
  const int*   act_ids = (const int*)d_in[1];
  const int*   obs_ids = (const int*)d_in[2];
  const float* res     = (const float*)d_in[3];
  const int*   lens    = (const int*)d_in[4];
  const float* act_emb = (const float*)d_in[5];
  const float* obs_emb = (const float*)d_in[6];
  const float* Wres    = (const float*)d_in[7];
  const float* bres    = (const float*)d_in[8];
  const float* WiF     = (const float*)d_in[9];
  const float* WhF     = (const float*)d_in[10];
  const float* biF     = (const float*)d_in[11];
  const float* bhF     = (const float*)d_in[12];
  const float* WiB     = (const float*)d_in[13];
  const float* WhB     = (const float*)d_in[14];
  const float* biB     = (const float*)d_in[15];
  const float* bhB     = (const float*)d_in[16];

  float* out     = (float*)d_out;            // [B,T,2H]
  float* out_h   = out + 16777216;           // [1,B,2H]
  float* out_c   = out_h + 65536;            // [1,B,2H]
  float* out_emb = out_c + 65536;            // [B,T,512]

  unsigned char* ws = (unsigned char*)d_ws;
  unsigned* cnt  = (unsigned*)(ws + WS_CNT);
  bf16* whf_bf   = (bf16*)(ws + WS_WHF);
  bf16* whb_bf   = (bf16*)(ws + WS_WHB);
  bf16* wif_bf   = (bf16*)(ws + WS_WIF);
  bf16* wib_bf   = (bf16*)(ws + WS_WIB);
  bf16* wres_bf  = (bf16*)(ws + WS_WRES);
  bf16* hbf      = (bf16*)(ws + WS_HBF);
  bf16* emb_bf   = (bf16*)(ws + WS_EMBBF);
  float* xwf     = (float*)(ws + WS_XWF);
  float* xwb     = (float*)(ws + WS_XWB);

  k_init <<<4096, 256, 0, stream>>>(out, WhF, WhB, WiF, WiB, Wres, ws);
  k_embed<<<4096, 256, 0, stream>>>(mask, act_ids, obs_ids, act_emb, obs_emb,
                                    out_emb, emb_bf);
  k_res  <<<512, 256, 0, stream>>>(res, wres_bf, bres, mask, out_emb, emb_bf);
  k_xw   <<<4096, 256, 0, stream>>>(emb_bf, wif_bf, biF, bhF, lens, xwf, 0);
  k_xw   <<<4096, 256, 0, stream>>>(emb_bf, wib_bf, biB, bhB, lens, xwb, 1);
  k_lstm <<<64, 256, 0, stream>>>(whf_bf, whb_bf, xwf, xwb, lens, hbf, out,
                                  out_h, out_c, cnt);
}
